// DDiTBlock_2576980377982
// MI455X (gfx1250) — matrix operand order repaired and checked by reading, not compiled, not run on hardware
//
#include <hip/hip_runtime.h>
#include <stdint.h>

// ---------------------------------------------------------------------------
// DiT block for MI455X (gfx1250): bf16 WMMA GEMMs + flash attention,
// async global->LDS staging (double buffered), BK=64 / 64-key tiles.
// B=2, S=2048, D=1024, H=16, HD=64, COND=1024, MLP_H=4096
// ---------------------------------------------------------------------------

typedef __bf16 bf16_t;
typedef __attribute__((ext_vector_type(16))) __bf16 v16bf;
typedef __attribute__((ext_vector_type(8)))  float  v8f;

#define TOK   4096
#define S_LEN 2048
#define DIM   1024
#define NH    16
#define HDIM  64
#define MLPH  4096
#define LN_EPS 1e-6f

union BFrag { v16bf v; unsigned u[8]; };

static __device__ __forceinline__ bf16_t f2bf(float f) {
  unsigned u = __builtin_bit_cast(unsigned, f);
  unsigned r = (u + 0x7fffu + ((u >> 16) & 1u)) >> 16;
  unsigned short h = (unsigned short)r;
  return __builtin_bit_cast(bf16_t, h);
}
static __device__ __forceinline__ v8f vzero8() {
  v8f z; for (int i = 0; i < 8; ++i) z[i] = 0.0f; return z;
}
// async copy of 16 bytes: global -> LDS (no VGPR round trip, ASYNCcnt tracked)
static __device__ __forceinline__ void async_cp16(unsigned ldsOff, const void* gaddr) {
  asm volatile("global_load_async_to_lds_b128 %0, %1, off"
               :: "v"(ldsOff), "v"(gaddr) : "memory");
}
static __device__ __forceinline__ void async_wait0() {
  asm volatile("s_wait_asynccnt 0" ::: "memory");
}

// ---------------------------------------------------------------------------
// mod = c @ W_ada + b_ada   (2 x 6144, K=1024) — tiny, plain VALU
// ---------------------------------------------------------------------------
__global__ void k_mod(const float* __restrict__ c, const float* __restrict__ Wada,
                      const float* __restrict__ bada, float* __restrict__ mod) {
  int j = blockIdx.x * blockDim.x + threadIdx.x;
  if (j >= 2 * 6 * DIM) return;
  int b = j / (6 * DIM), col = j % (6 * DIM);
  float acc = bada[col];
  const float* crow = c + (size_t)b * 1024;
  for (int k = 0; k < 1024; ++k)
    acc = fmaf(crow[k], Wada[(size_t)k * 6144 + col], acc);
  mod[j] = acc;
}

// ---------------------------------------------------------------------------
// Tiled transpose + f32->bf16: dst[n*K+k] = bf16(src[k*N+n]).
// block (32,8), grid (N/32, K/32).
// ---------------------------------------------------------------------------
__global__ void k_cvt_t(const float* __restrict__ src, bf16_t* __restrict__ dst,
                        int K, int N) {
  __shared__ float tile[32][33];
  int n0 = blockIdx.x * 32, k0 = blockIdx.y * 32;
  for (int i = 0; i < 4; ++i) {
    int k = k0 + threadIdx.y + i * 8;
    tile[threadIdx.y + i * 8][threadIdx.x] = src[(size_t)k * N + n0 + threadIdx.x];
  }
  __syncthreads();
  for (int i = 0; i < 4; ++i) {
    int n = n0 + threadIdx.y + i * 8;
    dst[(size_t)n * K + k0 + threadIdx.x] = f2bf(tile[threadIdx.x][threadIdx.y + i * 8]);
  }
}

// ---------------------------------------------------------------------------
// LayerNorm + adaLN modulate -> bf16.  One block per token row (D=1024).
// ---------------------------------------------------------------------------
__global__ void k_ln_mod(const float* __restrict__ x, const float* __restrict__ g,
                         const float* __restrict__ bta, const float* __restrict__ mod,
                         int shChunk, int scChunk, bf16_t* __restrict__ out) {
  int row = blockIdx.x;                 // token
  int b = row / S_LEN;
  const float* xr = x + (size_t)row * DIM;
  float s1 = 0.f, s2 = 0.f;
  float vals[4];
  for (int i = 0; i < 4; ++i) {
    int d = threadIdx.x + i * 256;
    float v = xr[d];
    vals[i] = v; s1 += v; s2 += v * v;
  }
  for (int m = 1; m < 32; m <<= 1) { s1 += __shfl_xor(s1, m, 32); s2 += __shfl_xor(s2, m, 32); }
  __shared__ float sh1[8], sh2[8];
  int wid = threadIdx.x >> 5, lane = threadIdx.x & 31;
  if (lane == 0) { sh1[wid] = s1; sh2[wid] = s2; }
  __syncthreads();
  float t1 = 0.f, t2 = 0.f;
  for (int i = 0; i < 8; ++i) { t1 += sh1[i]; t2 += sh2[i]; }
  float mean = t1 * (1.0f / DIM);
  float var  = t2 * (1.0f / DIM) - mean * mean;
  float rstd = rsqrtf(var + LN_EPS);
  const float* shp = mod + (size_t)b * 6 * DIM + shChunk * DIM;
  const float* scp = mod + (size_t)b * 6 * DIM + scChunk * DIM;
  bf16_t* orow = out + (size_t)row * DIM;
  for (int i = 0; i < 4; ++i) {
    int d = threadIdx.x + i * 256;
    float xn = (vals[i] - mean) * rstd * g[d] + bta[d];
    orow[d] = f2bf(xn * (1.0f + scp[d]) + shp[d]);
  }
}

// ---------------------------------------------------------------------------
// bf16 GEMM: C(f32)[M,N] = A[M,K] @ Bt[N,K]^T  (Bt is pre-transposed weights).
// Block 256 threads (8 waves), tile 128x128, BK=64, double-buffered async LDS.
// Wave (2x4 grid) computes 64x32 via 4x2 WMMAs per 32-wide k-step (16/tile).
// ---------------------------------------------------------------------------
__global__ __launch_bounds__(256)
void k_gemm_bf16(const bf16_t* __restrict__ A, const bf16_t* __restrict__ Bt,
                 float* __restrict__ C, int M, int N, int K) {
  __shared__ __align__(16) bf16_t As[2][128 * 64];   // [m][k], 16KB per buffer
  __shared__ __align__(16) bf16_t Bs[2][128 * 64];   // [n][k]
  const int tid = threadIdx.x;
  const int lane = tid & 31, wid = tid >> 5;
  const int l16 = lane & 15, hi = lane >> 4;
  const int waveM = wid >> 2, waveN = wid & 3;       // 2 x 4
  const int blockM = blockIdx.y * 128, blockN = blockIdx.x * 128;

  const unsigned asBase = (unsigned)(uintptr_t)(&As[0][0]);
  const unsigned bsBase = (unsigned)(uintptr_t)(&Bs[0][0]);

  // stage one 128x64 A tile and 128x64 B tile: 16KB each, 16B chunks per lane
  auto stage = [&](int buf, int kt) {
    for (int i = 0; i < 4; ++i) {
      int c = tid * 4 + i;                 // 0..1023 chunks
      int byteoff = c * 16;                // 0..16383
      int row = byteoff >> 7;              // 128B per tile row
      int within = byteoff & 127;
      const char* ga = (const char*)A + ((size_t)(blockM + row) * K + kt) * 2 + within;
      async_cp16(asBase + buf * 16384 + byteoff, ga);
      const char* gb = (const char*)Bt + ((size_t)(blockN + row) * K + kt) * 2 + within;
      async_cp16(bsBase + buf * 16384 + byteoff, gb);
    }
  };

  v8f acc[4][2];
  for (int i = 0; i < 4; ++i) for (int j = 0; j < 2; ++j) acc[i][j] = vzero8();

  const int nk = K / 64;
  stage(0, 0);
  async_wait0();
  __syncthreads();

  for (int t = 0; t < nk; ++t) {
    const int buf = t & 1;
    if (t + 1 < nk) stage(buf ^ 1, (t + 1) * 64);   // overlap with compute

    const bf16_t* Asb = &As[buf][0];
    const bf16_t* Bsb = &Bs[buf][0];
#pragma unroll
    for (int ks = 0; ks < 2; ++ks) {
      BFrag a[4], bfr[2];
      for (int i = 0; i < 4; ++i) {
        int m = waveM * 64 + i * 16 + l16;
        for (int j = 0; j < 8; ++j) {
          int k0 = ((j >> 2) << 4) + (hi << 3) + ((2 * j) & 7);   // ISA 16-bit A layout
          a[i].u[j] = *reinterpret_cast<const unsigned*>(&Asb[m * 64 + ks * 32 + k0]);
        }
      }
      for (int i = 0; i < 2; ++i) {
        int n = waveN * 32 + i * 16 + l16;
        for (int j = 0; j < 8; ++j) {
          int k0 = ((j >> 2) << 4) + (hi << 3) + ((2 * j) & 7);   // ISA 16-bit B layout
          bfr[i].u[j] = *reinterpret_cast<const unsigned*>(&Bsb[n * 64 + ks * 32 + k0]);
        }
      }
      for (int i = 0; i < 4; ++i)
        for (int jn = 0; jn < 2; ++jn)
          acc[i][jn] = __builtin_amdgcn_wmma_f32_16x16x32_bf16(
              false, a[i].v, false, bfr[jn].v, (short)0, acc[i][jn], false, false);
    }

    async_wait0();      // next tile landed in the other buffer
    __syncthreads();    // everyone done reading current buffer
  }

  // epilogue: C/D layout row = v + 8*hi, col = l16
  for (int i = 0; i < 4; ++i)
    for (int jn = 0; jn < 2; ++jn) {
      int col = blockN + waveN * 32 + jn * 16 + l16;
      for (int v = 0; v < 8; ++v) {
        int row = blockM + waveM * 64 + i * 16 + v + hi * 8;
        C[(size_t)row * N + col] = acc[i][jn][v];
      }
    }
}

// ---------------------------------------------------------------------------
// RoPE (applied to q, k AND v per the reference) + f32->bf16 split.
// q,k kept [B,S,H,HD]; v written transposed as vT[B,H,HD,S] for attention.
// ---------------------------------------------------------------------------
__global__ void k_rope(const float* __restrict__ qkv, const float* __restrict__ cosT,
                       const float* __restrict__ sinT,
                       bf16_t* __restrict__ qb, bf16_t* __restrict__ kb,
                       bf16_t* __restrict__ vtb) {
  int idx = blockIdx.x * blockDim.x + threadIdx.x;
  const int TOTAL = TOK * 3 * NH * (HDIM / 2);
  if (idx >= TOTAL) return;
  int t = idx / (3 * NH * 32);
  int r = idx % (3 * NH * 32);
  int which = r / (NH * 32);
  int r2 = r % (NH * 32);
  int h = r2 / 32, i = r2 % 32;
  int b = t / S_LEN, s = t % S_LEN;
  const float* src = qkv + (size_t)t * (3 * DIM) + which * DIM + h * HDIM;
  float x0 = src[2 * i], y0 = src[2 * i + 1];
  float cc = cosT[(size_t)s * HDIM + i];
  float sn = sinT[(size_t)s * HDIM + i];
  float o0 = x0 * cc - y0 * sn;
  float o1 = x0 * sn + y0 * cc;
  if (which == 2) {
    bf16_t* dst = vtb + ((size_t)(b * NH + h) * HDIM) * S_LEN;
    dst[(size_t)(2 * i) * S_LEN + s]     = f2bf(o0);
    dst[(size_t)(2 * i + 1) * S_LEN + s] = f2bf(o1);
  } else {
    bf16_t* dst = (which == 0 ? qb : kb) + (size_t)t * DIM + h * HDIM;
    dst[2 * i]     = f2bf(o0);
    dst[2 * i + 1] = f2bf(o1);
  }
}

// ---------------------------------------------------------------------------
// Flash attention.  grid = (S/64, H, B), block = 128 (4 waves).
// Each wave: 16 queries, online softmax over 32 steps of 64 keys.
// k tile [key][hd] and vT tile [hd][key] staged via async copies, 2 buffers.
// Per iteration: 8 WMMAs for scores (4 key-frags x 2 hd-ksteps) +
//                8 WMMAs for P@V (2 key-ksteps x 4 hd-frags).
// ---------------------------------------------------------------------------
__global__ __launch_bounds__(128)
void k_attn(const bf16_t* __restrict__ qbuf, const bf16_t* __restrict__ kbuf,
            const bf16_t* __restrict__ vT, bf16_t* __restrict__ obuf) {
  __shared__ __align__(16) bf16_t kt_s[2][64 * HDIM];   // [key][hd]  8KB/buf
  __shared__ __align__(16) bf16_t vt_s[2][HDIM * 64];   // [hd][key]  8KB/buf
  __shared__ __align__(16) bf16_t pbuf[4][16 * 64];     // per-wave probs 2KB

  const int tid = threadIdx.x;
  const int lane = tid & 31, w = tid >> 5;
  const int l16 = lane & 15, hi = lane >> 4;
  const int h = blockIdx.y, b = blockIdx.z;
  const int qBase = blockIdx.x * 64 + w * 16;

  const bf16_t* qb  = qbuf + ((size_t)b * S_LEN) * DIM + h * HDIM;
  const bf16_t* kb  = kbuf + ((size_t)b * S_LEN) * DIM + h * HDIM;
  const bf16_t* vtb = vT + ((size_t)(b * NH + h) * HDIM) * S_LEN;

  const unsigned ktBase = (unsigned)(uintptr_t)(&kt_s[0][0]);
  const unsigned vtBase = (unsigned)(uintptr_t)(&vt_s[0][0]);

  auto stageKV = [&](int buf, int keyBase) {
    for (int i = 0; i < 4; ++i) {
      int c = tid * 4 + i;            // 0..511 chunks of 16B (8KB per tile)
      int byteoff = c * 16;
      int krow = byteoff >> 7;        // key rows are 128B (64 x bf16)
      int kwithin = byteoff & 127;
      const char* gk = (const char*)kb + ((size_t)(keyBase + krow) * DIM) * 2 + kwithin;
      async_cp16(ktBase + buf * 8192 + byteoff, gk);
      // vT rows are 128B too (64 keys x bf16)
      const char* gv = (const char*)vtb + ((size_t)krow * S_LEN + keyBase) * 2 + kwithin;
      async_cp16(vtBase + buf * 8192 + byteoff, gv);
    }
  };

  // q A-fragments once (K = HD = 64 -> 2 k-steps)
  BFrag aq[2];
  for (int ks = 0; ks < 2; ++ks)
    for (int j = 0; j < 8; ++j) {
      int k0 = ((j >> 2) << 4) + (hi << 3) + ((2 * j) & 7);
      int hd = ks * 32 + k0;
      aq[ks].u[j] = *reinterpret_cast<const unsigned*>(&qb[(size_t)(qBase + l16) * DIM + hd]);
    }

  v8f oacc[4];
  for (int nf = 0; nf < 4; ++nf) oacc[nf] = vzero8();
  float m8[8], l8[8];
  for (int v = 0; v < 8; ++v) { m8[v] = -1e30f; l8[v] = 0.0f; }
  const float scale = 0.125f;   // 1/sqrt(64)

  const int nIt = S_LEN / 64;
  stageKV(0, 0);
  async_wait0();
  __syncthreads();

  for (int it = 0; it < nIt; ++it) {
    const int buf = it & 1;
    if (it + 1 < nIt) stageKV(buf ^ 1, (it + 1) * 64);

    const bf16_t* kts = &kt_s[buf][0];
    const bf16_t* vts = &vt_s[buf][0];

    // scores = q @ k^T : 4 key-frags x 2 hd k-steps
    v8f sc[4];
    for (int nf = 0; nf < 4; ++nf) sc[nf] = vzero8();
#pragma unroll
    for (int ks = 0; ks < 2; ++ks)
      for (int nf = 0; nf < 4; ++nf) {
        BFrag bk;
        for (int j = 0; j < 8; ++j) {
          int k0 = ((j >> 2) << 4) + (hi << 3) + ((2 * j) & 7);
          bk.u[j] = *reinterpret_cast<const unsigned*>(&kts[(nf * 16 + l16) * HDIM + ks * 32 + k0]);
        }
        sc[nf] = __builtin_amdgcn_wmma_f32_16x16x32_bf16(
            false, aq[ks].v, false, bk.v, (short)0, sc[nf], false, false);
      }

    // online softmax (row = v + 8*hi; 16-lane group reductions)
    for (int v = 0; v < 8; ++v) {
      float xv[4];
      float mx = -1e30f;
      for (int nf = 0; nf < 4; ++nf) { xv[nf] = sc[nf][v] * scale; mx = fmaxf(mx, xv[nf]); }
      for (int msk = 1; msk < 16; msk <<= 1) mx = fmaxf(mx, __shfl_xor(mx, msk, 16));
      float mnew = fmaxf(m8[v], mx);
      float corr = __expf(m8[v] - mnew);
      float rs = 0.f;
      int m = v + hi * 8;
      for (int nf = 0; nf < 4; ++nf) {
        float p = __expf(xv[nf] - mnew);
        rs += p;
        pbuf[w][m * 64 + nf * 16 + l16] = f2bf(p);
      }
      for (int msk = 1; msk < 16; msk <<= 1) rs += __shfl_xor(rs, msk, 16);
      l8[v] = l8[v] * corr + rs;
      m8[v] = mnew;
      for (int nf = 0; nf < 4; ++nf) oacc[nf][v] *= corr;
    }
    // pbuf is private to this wave; same-wave LDS ops are in order.

    // O += P(16x64) @ V(64x64): 2 key k-steps x 4 hd-frags
#pragma unroll
    for (int ks = 0; ks < 2; ++ks) {
      BFrag ap;
      for (int j = 0; j < 8; ++j) {
        int k0 = ((j >> 2) << 4) + (hi << 3) + ((2 * j) & 7);
        ap.u[j] = *reinterpret_cast<const unsigned*>(&pbuf[w][l16 * 64 + ks * 32 + k0]);
      }
      for (int nf = 0; nf < 4; ++nf) {
        BFrag bv;
        for (int j = 0; j < 8; ++j) {
          int k0 = ((j >> 2) << 4) + (hi << 3) + ((2 * j) & 7);
          bv.u[j] = *reinterpret_cast<const unsigned*>(&vts[(nf * 16 + l16) * 64 + ks * 32 + k0]);
        }
        oacc[nf] = __builtin_amdgcn_wmma_f32_16x16x32_bf16(
            false, ap.v, false, bv.v, (short)0, oacc[nf], false, false);
      }
    }

    async_wait0();
    __syncthreads();
  }

  // normalize and write bf16 output [B,S,H,HD]
  for (int v = 0; v < 8; ++v) {
    float linv = 1.0f / l8[v];
    int srow = qBase + v + hi * 8;
    for (int nf = 0; nf < 4; ++nf) {
      int hd = nf * 16 + l16;
      obuf[((size_t)b * S_LEN + srow) * DIM + h * HDIM + hd] = f2bf(oacc[nf][v] * linv);
    }
  }
}

// ---------------------------------------------------------------------------
// Epilogues
// ---------------------------------------------------------------------------
__global__ void k_resid_gate(const float* __restrict__ xin, const float* __restrict__ y,
                             const float* __restrict__ mod, int gateChunk,
                             float* __restrict__ out) {
  int idx = blockIdx.x * blockDim.x + threadIdx.x;
  if (idx >= TOK * DIM) return;
  int t = idx / DIM, d = idx % DIM;
  int b = t / S_LEN;
  float g = mod[(size_t)b * 6 * DIM + gateChunk * DIM + d];
  out[idx] = xin[idx] + y[idx] * g;
}

__global__ void k_resid_gate_bias(const float* __restrict__ xin, const float* __restrict__ y,
                                  const float* __restrict__ bias, const float* __restrict__ mod,
                                  int gateChunk, float* __restrict__ out) {
  int idx = blockIdx.x * blockDim.x + threadIdx.x;
  if (idx >= TOK * DIM) return;
  int t = idx / DIM, d = idx % DIM;
  int b = t / S_LEN;
  float g = mod[(size_t)b * 6 * DIM + gateChunk * DIM + d];
  out[idx] = xin[idx] + (y[idx] + bias[d]) * g;
}

__global__ void k_gelu_bias(const float* __restrict__ y, const float* __restrict__ bias,
                            bf16_t* __restrict__ out, int n, int ncols) {
  int idx = blockIdx.x * blockDim.x + threadIdx.x;
  if (idx >= n) return;
  float x = y[idx] + bias[idx % ncols];
  float t = tanhf(0.7978845608028654f * (x + 0.044715f * x * x * x));
  out[idx] = f2bf(0.5f * x * (1.0f + t));
}

// ---------------------------------------------------------------------------
// launch
// ---------------------------------------------------------------------------
extern "C" void kernel_launch(void* const* d_in, const int* in_sizes, int n_in,
                              void* d_out, int out_size, void* d_ws, size_t ws_size,
                              hipStream_t stream) {
  (void)in_sizes; (void)n_in; (void)out_size; (void)ws_size;
  const float* x     = (const float*)d_in[0];
  const float* c     = (const float*)d_in[1];
  const float* cosT  = (const float*)d_in[2];
  const float* sinT  = (const float*)d_in[3];
  const float* ln1_g = (const float*)d_in[4];
  const float* ln1_b = (const float*)d_in[5];
  const float* ln2_g = (const float*)d_in[6];
  const float* ln2_b = (const float*)d_in[7];
  const float* W_qkv = (const float*)d_in[8];
  const float* W_out = (const float*)d_in[9];
  const float* W1    = (const float*)d_in[10];
  const float* b1    = (const float*)d_in[11];
  const float* W2    = (const float*)d_in[12];
  const float* b2    = (const float*)d_in[13];
  const float* W_ada = (const float*)d_in[14];
  const float* b_ada = (const float*)d_in[15];
  float* out = (float*)d_out;

  char* ws = (char*)d_ws;
  size_t off = 0;
  auto alloc = [&](size_t bytes) -> char* {
    char* p = ws + off;
    off = (off + bytes + 255) & ~(size_t)255;
    return p;
  };
  float*  modw     = (float*)alloc(2 * 6 * DIM * sizeof(float));
  bf16_t* wqkvT_bf = (bf16_t*)alloc((size_t)DIM * 3 * DIM * 2);   // [3D, D]
  bf16_t* woutT_bf = (bf16_t*)alloc((size_t)DIM * DIM * 2);       // [D, D]
  bf16_t* w1T_bf   = (bf16_t*)alloc((size_t)DIM * MLPH * 2);      // [MLPH, D]
  bf16_t* w2T_bf   = (bf16_t*)alloc((size_t)MLPH * DIM * 2);      // [D, MLPH]
  bf16_t* xn_bf    = (bf16_t*)alloc((size_t)TOK * DIM * 2);
  float*  gemmo    = (float*)alloc((size_t)TOK * MLPH * sizeof(float));  // reused
  bf16_t* q_bf     = (bf16_t*)alloc((size_t)TOK * DIM * 2);
  bf16_t* k_bf     = (bf16_t*)alloc((size_t)TOK * DIM * 2);
  bf16_t* vT_bf    = (bf16_t*)alloc((size_t)TOK * DIM * 2);       // [B,H,HD,S]
  bf16_t* attn_bf  = (bf16_t*)alloc((size_t)TOK * DIM * 2);
  float*  xmid     = (float*)alloc((size_t)TOK * DIM * sizeof(float));
  bf16_t* h_bf     = (bf16_t*)alloc((size_t)TOK * MLPH * 2);

  // 1. adaLN modulation vector
  k_mod<<<(2 * 6 * DIM + 255) / 256, 256, 0, stream>>>(c, W_ada, b_ada, modw);

  // 2. weight transpose+convert f32 -> bf16 [N,K]
  k_cvt_t<<<dim3(3 * DIM / 32, DIM / 32), dim3(32, 8), 0, stream>>>(W_qkv, wqkvT_bf, DIM, 3 * DIM);
  k_cvt_t<<<dim3(DIM / 32, DIM / 32),     dim3(32, 8), 0, stream>>>(W_out, woutT_bf, DIM, DIM);
  k_cvt_t<<<dim3(MLPH / 32, DIM / 32),    dim3(32, 8), 0, stream>>>(W1, w1T_bf, DIM, MLPH);
  k_cvt_t<<<dim3(DIM / 32, MLPH / 32),    dim3(32, 8), 0, stream>>>(W2, w2T_bf, MLPH, DIM);

  // 3. LN1 + modulate (sh=chunk0, sc=chunk1)
  k_ln_mod<<<TOK, 256, 0, stream>>>(x, ln1_g, ln1_b, modw, 0, 1, xn_bf);

  // 4. QKV GEMM: [4096,1024] @ [1024,3072]
  k_gemm_bf16<<<dim3(3 * DIM / 128, TOK / 128), 256, 0, stream>>>(xn_bf, wqkvT_bf, gemmo, TOK, 3 * DIM, DIM);

  // 5. RoPE + split to bf16 q/k (row-major) and vT (transposed)
  k_rope<<<(TOK * 3 * NH * 32 + 255) / 256, 256, 0, stream>>>(gemmo, cosT, sinT, q_bf, k_bf, vT_bf);

  // 6. flash attention
  k_attn<<<dim3(S_LEN / 64, NH, 2), 128, 0, stream>>>(q_bf, k_bf, vT_bf, attn_bf);

  // 7. output projection: [4096,1024] @ [1024,1024]
  k_gemm_bf16<<<dim3(DIM / 128, TOK / 128), 256, 0, stream>>>(attn_bf, woutT_bf, gemmo, TOK, DIM, DIM);

  // 8. gated residual (g_msa = chunk2)
  k_resid_gate<<<(TOK * DIM + 255) / 256, 256, 0, stream>>>(x, gemmo, modw, 2, xmid);

  // 9. LN2 + modulate (sh=chunk3, sc=chunk4)
  k_ln_mod<<<TOK, 256, 0, stream>>>(xmid, ln2_g, ln2_b, modw, 3, 4, xn_bf);

  // 10. MLP up: [4096,1024] @ [1024,4096]
  k_gemm_bf16<<<dim3(MLPH / 128, TOK / 128), 256, 0, stream>>>(xn_bf, w1T_bf, gemmo, TOK, MLPH, DIM);

  // 11. bias + GELU -> bf16
  k_gelu_bias<<<((size_t)TOK * MLPH + 255) / 256, 256, 0, stream>>>(gemmo, b1, h_bf, TOK * MLPH, MLPH);

  // 12. MLP down: [4096,4096] @ [4096,1024]
  k_gemm_bf16<<<dim3(DIM / 128, TOK / 128), 256, 0, stream>>>(h_bf, w2T_bf, gemmo, TOK, DIM, MLPH);

  // 13. final gated residual with bias (g_mlp = chunk5)
  k_resid_gate_bias<<<(TOK * DIM + 255) / 256, 256, 0, stream>>>(xmid, gemmo, b2, modw, 5, out);
}
